// Encoder_29832842838074
// MI455X (gfx1250) — compile-verified, matrix operand order checked by reading, not run on hardware
//
#include <hip/hip_runtime.h>
#include <cstddef>

// ---------------------------------------------------------------------------
// Types for CDNA5 WMMA (wave32, 16x16x32 bf16 -> f32)
// ---------------------------------------------------------------------------
typedef __attribute__((ext_vector_type(16))) __bf16 v16bf;
typedef __attribute__((ext_vector_type(8)))  __bf16 v8bf;
typedef __attribute__((ext_vector_type(8)))  float  v8f;
typedef __attribute__((ext_vector_type(4)))  int    v4i;

#define WMMA_BF16(a, b, c) \
  __builtin_amdgcn_wmma_f32_16x16x32_bf16(false, (a), false, (b), (short)0, (c), false, false)

// Async global->LDS (CDNA5, tracked with ASYNCcnt). Guarded: toolchains differ.
#if defined(__has_builtin)
#if __has_builtin(__builtin_amdgcn_global_load_async_to_lds_b128)
#define HAVE_ASYNC_LDS 1
#endif
#endif

// Problem constants
#define BB     4
#define NCAM   6
#define DMODEL 128
#define NHEAD  4
#define DH     32
#define QLEN   1024   // 32*32
#define KLEN   1344   // 24*56
#define NKTOT  8064   // NCAM*KLEN
#define SCALE  0.17677669529663687f

// Load an A/B WMMA operand (16x32 bf16 tile) for this lane.
// Per ISA layout: elements 0..7  = dims [off, off+8)
//                 elements 8..15 = dims [16+off, 16+off+8), off = 8*(lane>=16).
__device__ inline v16bf ld_ab(const __bf16* p0, const __bf16* p1) {
  v8bf x = *(const v8bf*)p0;
  v8bf y = *(const v8bf*)p1;
  v16bf r;
#pragma unroll
  for (int i = 0; i < 8; ++i) { r[i] = x[i]; r[i + 8] = y[i]; }
  return r;
}

// ---------------------------------------------------------------------------
// Weight transpose + fp32 -> bf16:  W[k][c] -> Wt[c][k]
// ---------------------------------------------------------------------------
__global__ __launch_bounds__(256) void wtrans_kernel(const float* __restrict__ W,
                                                     __bf16* __restrict__ Wt,
                                                     int Kdim, int NC) {
  int idx = blockIdx.x * blockDim.x + threadIdx.x;
  if (idx >= Kdim * NC) return;
  int c = idx / Kdim, k = idx % Kdim;
  Wt[idx] = (__bf16)W[(size_t)k * NC + c];
}

// ---------------------------------------------------------------------------
// LayerNorm over channel dim of (bn, D, S) fp32 input -> bf16 [row][128]
// One lane owns one row (spatial position): coalesced d-loop loads.
// ---------------------------------------------------------------------------
__global__ __launch_bounds__(256) void ln_chw_kernel(const float* __restrict__ x,
                                                     const float* __restrict__ g,
                                                     const float* __restrict__ bt,
                                                     __bf16* __restrict__ out,
                                                     int S, int totalRows) {
  int lane = threadIdx.x & 31;
  int wv = blockIdx.x * (blockDim.x >> 5) + (threadIdx.x >> 5);
  int row = wv * 32 + lane;
  if (row >= totalRows) return;
  int bn = row / S, s = row % S;
  const float* base = x + (size_t)bn * DMODEL * S + s;
  float sum = 0.f, sq = 0.f;
  for (int d = 0; d < DMODEL; ++d) {
    float v = base[(size_t)d * S];
    sum += v; sq += v * v;
  }
  float mean = sum * (1.f / DMODEL);
  float var  = sq * (1.f / DMODEL) - mean * mean;
  float rstd = rsqrtf(var + 1e-5f);
  __bf16* o = out + (size_t)row * DMODEL;
  for (int d = 0; d < DMODEL; ++d) {
    float v = base[(size_t)d * S];
    o[d] = (__bf16)(((v - mean) * rstd) * g[d] + bt[d]);
  }
}

// ---------------------------------------------------------------------------
// LayerNorm over rows of fp32 [M][128]; optional bf16 and/or fp32 output,
// optional channels-first transposed fp32 output (final result).
// ---------------------------------------------------------------------------
__global__ __launch_bounds__(256) void ln_rows_kernel(const float* __restrict__ z,
                                                      const float* __restrict__ g,
                                                      const float* __restrict__ bt,
                                                      __bf16* __restrict__ out_bf,
                                                      float* __restrict__ out_f,
                                                      int M, int transpose_out) {
  int lane = threadIdx.x & 31;
  int row = blockIdx.x * (blockDim.x >> 5) + (threadIdx.x >> 5);
  if (row >= M) return;
  const float* zr = z + (size_t)row * DMODEL;
  float v[4];
  float sum = 0.f, sq = 0.f;
#pragma unroll
  for (int i = 0; i < 4; ++i) {
    v[i] = zr[lane + 32 * i];
    sum += v[i]; sq += v[i] * v[i];
  }
#pragma unroll
  for (int m = 16; m >= 1; m >>= 1) {
    sum += __shfl_xor(sum, m);
    sq  += __shfl_xor(sq, m);
  }
  float mean = sum * (1.f / DMODEL);
  float var  = sq * (1.f / DMODEL) - mean * mean;
  float rstd = rsqrtf(var + 1e-5f);
#pragma unroll
  for (int i = 0; i < 4; ++i) {
    int d = lane + 32 * i;
    float y = (v[i] - mean) * rstd * g[d] + bt[d];
    if (out_bf) out_bf[(size_t)row * DMODEL + d] = (__bf16)y;
    if (out_f) {
      if (transpose_out) {
        int b = row >> 10, s = row & 1023;  // Q = 1024
        out_f[((size_t)(b * DMODEL + d)) * QLEN + s] = y;
      } else {
        out_f[(size_t)row * DMODEL + d] = y;
      }
    }
  }
}

// ---------------------------------------------------------------------------
// Generic WMMA GEMM:  C[M,NC] = A[M,Kdim](bf16) x Wt[NC,Kdim](bf16)
//                     + bias, optional exact GELU, epilogue modes.
// One wave computes one 16x16 output tile; grid = (M/16, NC/128).
// mode 0: out_bf[row*NC+col]
// mode 1: V transpose -> out_bf[((b*4+h)*32+d)*8064 + key]   (rows = b,n*K+k)
// mode 2: out_f[row*128+col] = acc + skip[(b*128+col)*1024 + s]   (rows = b*Q+s)
// mode 3: out_f[row*128+col] = acc + add_f[row*128+col]
// ---------------------------------------------------------------------------
__global__ __launch_bounds__(256) void gemm_bf16_kernel(const __bf16* __restrict__ A,
                                                        const __bf16* __restrict__ Wt,
                                                        const float* __restrict__ bias,
                                                        int Kdim, int NC, int M,
                                                        int mode, int do_gelu,
                                                        __bf16* __restrict__ out_bf,
                                                        float* __restrict__ out_f,
                                                        const float* __restrict__ add_f) {
  int wave = threadIdx.x >> 5, lane = threadIdx.x & 31;
  int rowbase = blockIdx.x * 16;
  int colbase = blockIdx.y * 128 + wave * 16;
  int mrow = lane & 15;
  int off = (lane >> 4) * 8;
  int rowadd = (lane >> 4) * 8;

  v8f acc = {0.f, 0.f, 0.f, 0.f, 0.f, 0.f, 0.f, 0.f};
  const __bf16* arow = A + (size_t)(rowbase + mrow) * Kdim;
  const __bf16* brow = Wt + (size_t)(colbase + mrow) * Kdim;
  for (int kk = 0; kk < Kdim; kk += 32) {
    v16bf a = ld_ab(arow + kk + off, arow + kk + 16 + off);
    v16bf b = ld_ab(brow + kk + off, brow + kk + 16 + off);
    acc = WMMA_BF16(a, b, acc);
  }

#pragma unroll
  for (int r = 0; r < 8; ++r) {
    int row = rowbase + r + rowadd;
    int col = colbase + mrow;
    float val = acc[r] + bias[col];
    if (do_gelu) val = 0.5f * val * (1.f + erff(val * 0.70710678118654752f));
    if (mode == 0) {
      out_bf[(size_t)row * NC + col] = (__bf16)val;
    } else if (mode == 1) {
      int b = row / NKTOT;
      int key = row % NKTOT;
      out_bf[((size_t)((b * NHEAD + (col >> 5)) * DH + (col & 31))) * NKTOT + key] = (__bf16)val;
    } else if (mode == 2) {
      int b = row >> 10, s = row & 1023;
      out_f[(size_t)row * DMODEL + col] =
          val + add_f[((size_t)(b * DMODEL + col)) * QLEN + s];
    } else {
      out_f[(size_t)row * DMODEL + col] = val + add_f[(size_t)row * DMODEL + col];
    }
  }
}

// ---------------------------------------------------------------------------
// Flash attention. Block = 8 waves, all sharing one (b, head); wave w owns
// query tile (blk%8)*8 + w. Per 32-key chunk the whole block cooperatively
// stages the K tile (32 keys x 32 dims) and V tile (32 dh x 32 keys) into LDS
// via CDNA5 async global->LDS loads (ASYNCcnt), then every wave reads its
// WMMA operands from LDS -- an 8x cut in global K/V traffic vs per-wave loads.
//   qp : bf16 [b][n][q][128]     kp : bf16 [b][n][k][128]
//   vpT: bf16 [b][h][dh=32][key=8064]
//   out: bf16 a[b*Q][128] (heads merged)
// 1344 % 32 == 0 so a chunk never straddles a camera (queries are per-camera).
// ---------------------------------------------------------------------------
__global__ __launch_bounds__(256) void attn_kernel(const __bf16* __restrict__ qp,
                                                   const __bf16* __restrict__ kp,
                                                   const __bf16* __restrict__ vpT,
                                                   __bf16* __restrict__ aout) {
  __shared__ __align__(16) __bf16 Kt[32 * 32];      // [key][dim]   2KB
  __shared__ __align__(16) __bf16 Vt[32 * 32];      // [dh][key]    2KB
  __shared__ __align__(16) __bf16 Pl[8][16 * 32];   // per-wave P transpose slab
  int tid = threadIdx.x;
  int wave = tid >> 5, lane = tid & 31;
  int blk = blockIdx.x;                 // 128 blocks
  int bh = blk >> 3;
  int b = bh >> 2, h = bh & 3;
  int qbase = ((blk & 7) * 8 + wave) * 16;
  int mrow = lane & 15;
  int off = (lane >> 4) * 8;
  int rowadd = (lane >> 4) * 8;

  // staging role: threads 0..127 -> K tile, 128..255 -> V tile; 16B per thread
  int stgRow = (tid & 127) >> 2;        // 0..31
  int stgChunk = (tid & 3) * 8;         // element offset 0,8,16,24
  bool isV = tid >= 128;

  float m[8], l[8];
  v8f o0 = {0.f, 0.f, 0.f, 0.f, 0.f, 0.f, 0.f, 0.f};
  v8f o1 = {0.f, 0.f, 0.f, 0.f, 0.f, 0.f, 0.f, 0.f};
  v8f czero = {0.f, 0.f, 0.f, 0.f, 0.f, 0.f, 0.f, 0.f};
#pragma unroll
  for (int r = 0; r < 8; ++r) { m[r] = -1e30f; l[r] = 0.f; }

  __bf16* Pw = &Pl[wave][0];

  for (int n = 0; n < NCAM; ++n) {
    // Query operand for this camera (16 queries x 32 dims of head h)
    const __bf16* qrow =
        qp + ((size_t)((b * NCAM + n) * QLEN + qbase + mrow)) * DMODEL + h * DH;
    v16bf aq = ld_ab(qrow + off, qrow + 16 + off);
    const __bf16* kcam = kp + ((size_t)((b * NCAM + n) * KLEN)) * DMODEL + h * DH;
    const __bf16* vcam = vpT + ((size_t)((b * NHEAD + h) * DH)) * NKTOT + n * KLEN;

    for (int kc = 0; kc < KLEN; kc += 32) {
      // ---- cooperative stage of K/V tiles into LDS ----
      const __bf16* gsrc;
      __bf16* ldst;
      if (!isV) {
        gsrc = kcam + (size_t)(kc + stgRow) * DMODEL + stgChunk;
        ldst = &Kt[stgRow * 32 + stgChunk];
      } else {
        gsrc = vcam + (size_t)stgRow * NKTOT + kc + stgChunk;
        ldst = &Vt[stgRow * 32 + stgChunk];
      }
#ifdef HAVE_ASYNC_LDS
      __builtin_amdgcn_global_load_async_to_lds_b128(
          (__attribute__((address_space(1))) v4i*)gsrc,
          (__attribute__((address_space(3))) v4i*)ldst, 0, 0);
      asm volatile("s_wait_asynccnt 0" ::: "memory");
#else
      *(v8bf*)ldst = *(const v8bf*)gsrc;
#endif
      __syncthreads();

      // ---- scores: two 16x16 tiles over 32 keys (K operands from LDS) ----
      v16bf bk0 = ld_ab(&Kt[mrow * 32 + off], &Kt[mrow * 32 + 16 + off]);
      v16bf bk1 = ld_ab(&Kt[(16 + mrow) * 32 + off], &Kt[(16 + mrow) * 32 + 16 + off]);
      v8f s0 = WMMA_BF16(aq, bk0, czero);
      v8f s1 = WMMA_BF16(aq, bk1, czero);

      // ---- online softmax update per query row ----
#pragma unroll
      for (int r = 0; r < 8; ++r) {
        float a0 = s0[r] * SCALE, a1 = s1[r] * SCALE;
        float rm = fmaxf(a0, a1);
        rm = fmaxf(rm, __shfl_xor(rm, 1));
        rm = fmaxf(rm, __shfl_xor(rm, 2));
        rm = fmaxf(rm, __shfl_xor(rm, 4));
        rm = fmaxf(rm, __shfl_xor(rm, 8));
        float mn = fmaxf(m[r], rm);
        float alpha = __expf(m[r] - mn);
        float p0 = __expf(a0 - mn);
        float p1 = __expf(a1 - mn);
        float rs = p0 + p1;
        rs += __shfl_xor(rs, 1);
        rs += __shfl_xor(rs, 2);
        rs += __shfl_xor(rs, 4);
        rs += __shfl_xor(rs, 8);
        l[r] = l[r] * alpha + rs;
        m[r] = mn;
        o0[r] *= alpha;
        o1[r] *= alpha;
        int prow = r + rowadd;  // C-layout row of this vgpr/half
        Pw[prow * 32 + mrow] = (__bf16)p0;
        Pw[prow * 32 + 16 + mrow] = (__bf16)p1;
      }
      // LDS write -> read fence within the wave (CDNA5 split DS counter)
      asm volatile("s_wait_dscnt 0" ::: "memory");

      // ---- P (A-layout via LDS) x V (B operand from LDS V tile) ----
      const __bf16* pr = Pw + mrow * 32;
      v16bf ap = ld_ab(pr + off, pr + 16 + off);
      v16bf bv0 = ld_ab(&Vt[mrow * 32 + off], &Vt[mrow * 32 + 16 + off]);
      v16bf bv1 = ld_ab(&Vt[(16 + mrow) * 32 + off], &Vt[(16 + mrow) * 32 + 16 + off]);
      o0 = WMMA_BF16(ap, bv0, o0);
      o1 = WMMA_BF16(ap, bv1, o1);

      __syncthreads();  // all waves done with Kt/Vt before next stage
    }
  }

  // ---- normalize and write merged-head output ----
#pragma unroll
  for (int r = 0; r < 8; ++r) {
    int row = qbase + r + rowadd;
    float inv = 1.f / l[r];
    size_t o = ((size_t)(b * QLEN + row)) * DMODEL + h * DH;
    aout[o + mrow] = (__bf16)(o0[r] * inv);
    aout[o + 16 + mrow] = (__bf16)(o1[r] * inv);
  }
}

// ---------------------------------------------------------------------------
// Host orchestration
// ---------------------------------------------------------------------------
extern "C" void kernel_launch(void* const* d_in, const int* in_sizes, int n_in,
                              void* d_out, int out_size, void* d_ws, size_t ws_size,
                              hipStream_t stream) {
  (void)in_sizes; (void)n_in; (void)out_size; (void)ws_size;

  const float* q_in   = (const float*)d_in[0];
  const float* k_in   = (const float*)d_in[1];
  const float* v_in   = (const float*)d_in[2];
  const float* skip   = (const float*)d_in[3];
  const float* qn_g   = (const float*)d_in[4];
  const float* qn_b   = (const float*)d_in[5];
  const float* Wq     = (const float*)d_in[6];
  const float* bq     = (const float*)d_in[7];
  const float* kn_g   = (const float*)d_in[8];
  const float* kn_b   = (const float*)d_in[9];
  const float* Wk     = (const float*)d_in[10];
  const float* bk     = (const float*)d_in[11];
  const float* vn_g   = (const float*)d_in[12];
  const float* vn_b   = (const float*)d_in[13];
  const float* Wv     = (const float*)d_in[14];
  const float* bv     = (const float*)d_in[15];
  const float* Wp     = (const float*)d_in[16];
  const float* bp     = (const float*)d_in[17];
  const float* pre_g  = (const float*)d_in[18];
  const float* pre_b  = (const float*)d_in[19];
  const float* W1     = (const float*)d_in[20];
  const float* b1     = (const float*)d_in[21];
  const float* W2     = (const float*)d_in[22];
  const float* b2     = (const float*)d_in[23];
  const float* post_g = (const float*)d_in[24];
  const float* post_b = (const float*)d_in[25];
  float* out = (float*)d_out;

  const int MQ = BB * NCAM * QLEN;   // 24576 rows
  const int MK = BB * NCAM * KLEN;   // 32256 rows
  const int MZ = BB * QLEN;          // 4096 rows

  // Workspace carve-up (256B aligned)
  char* ws = (char*)d_ws;
  size_t off = 0;
  auto carve = [&](size_t bytes) {
    size_t o = off;
    off += (bytes + 255) & ~(size_t)255;
    return o;
  };
  __bf16* wqT = (__bf16*)(ws + carve(128 * 128 * 2));
  __bf16* wkT = (__bf16*)(ws + carve(128 * 128 * 2));
  __bf16* wvT = (__bf16*)(ws + carve(128 * 128 * 2));
  __bf16* wpT = (__bf16*)(ws + carve(128 * 128 * 2));
  __bf16* w1T = (__bf16*)(ws + carve(256 * 128 * 2));
  __bf16* w2T = (__bf16*)(ws + carve(128 * 256 * 2));
  __bf16* xq  = (__bf16*)(ws + carve((size_t)MQ * 128 * 2));
  __bf16* xk  = (__bf16*)(ws + carve((size_t)MK * 128 * 2));
  __bf16* xv  = (__bf16*)(ws + carve((size_t)MK * 128 * 2));
  __bf16* qp  = (__bf16*)(ws + carve((size_t)MQ * 128 * 2));
  __bf16* kp  = (__bf16*)(ws + carve((size_t)MK * 128 * 2));
  __bf16* vpT = (__bf16*)(ws + carve((size_t)BB * NHEAD * DH * NKTOT * 2));
  __bf16* abuf= (__bf16*)(ws + carve((size_t)MZ * 128 * 2));
  float*  zpre= (float*)(ws + carve((size_t)MZ * 128 * 4));
  float*  z1f = (float*)(ws + carve((size_t)MZ * 128 * 4));
  __bf16* z1b = (__bf16*)(ws + carve((size_t)MZ * 128 * 2));
  __bf16* h1  = (__bf16*)(ws + carve((size_t)MZ * 256 * 2));
  float*  z2  = (float*)(ws + carve((size_t)MZ * 128 * 4));

  // 1) bf16 transposed weights
  wtrans_kernel<<<64, 256, 0, stream>>>(Wq, wqT, 128, 128);
  wtrans_kernel<<<64, 256, 0, stream>>>(Wk, wkT, 128, 128);
  wtrans_kernel<<<64, 256, 0, stream>>>(Wv, wvT, 128, 128);
  wtrans_kernel<<<64, 256, 0, stream>>>(Wp, wpT, 128, 128);
  wtrans_kernel<<<128, 256, 0, stream>>>(W1, w1T, 128, 256);
  wtrans_kernel<<<128, 256, 0, stream>>>(W2, w2T, 256, 128);

  // 2) LayerNorms of q/k/v (channel-major input -> bf16 row-major)
  ln_chw_kernel<<<MQ / 256, 256, 0, stream>>>(q_in, qn_g, qn_b, xq, QLEN, MQ);
  ln_chw_kernel<<<MK / 256, 256, 0, stream>>>(k_in, kn_g, kn_b, xk, KLEN, MK);
  ln_chw_kernel<<<MK / 256, 256, 0, stream>>>(v_in, vn_g, vn_b, xv, KLEN, MK);

  // 3) Q/K/V projections (WMMA GEMMs)
  gemm_bf16_kernel<<<dim3(MQ / 16, 1), 256, 0, stream>>>(
      xq, wqT, bq, 128, 128, MQ, /*mode=*/0, /*gelu=*/0, qp, nullptr, nullptr);
  gemm_bf16_kernel<<<dim3(MK / 16, 1), 256, 0, stream>>>(
      xk, wkT, bk, 128, 128, MK, /*mode=*/0, /*gelu=*/0, kp, nullptr, nullptr);
  gemm_bf16_kernel<<<dim3(MK / 16, 1), 256, 0, stream>>>(
      xv, wvT, bv, 128, 128, MK, /*mode=*/1, /*gelu=*/0, vpT, nullptr, nullptr);

  // 4) Flash attention over all cameras (async-LDS staged K/V tiles)
  attn_kernel<<<BB * NHEAD * 8, 256, 0, stream>>>(qp, kp, vpT, abuf);

  // 5) Output projection + skip (fp32)
  gemm_bf16_kernel<<<dim3(MZ / 16, 1), 256, 0, stream>>>(
      abuf, wpT, bp, 128, 128, MZ, /*mode=*/2, /*gelu=*/0, nullptr, zpre, skip);

  // 6) pre-LN -> bf16 (GEMM input) + fp32 (residual)
  ln_rows_kernel<<<MZ / 8, 256, 0, stream>>>(zpre, pre_g, pre_b, z1b, z1f, MZ, 0);

  // 7) MLP: h = gelu(z1 @ W1 + b1)   (128 -> 256)
  gemm_bf16_kernel<<<dim3(MZ / 16, 2), 256, 0, stream>>>(
      z1b, w1T, b1, 128, 256, MZ, /*mode=*/0, /*gelu=*/1, h1, nullptr, nullptr);

  // 8) z2 = z1 + h @ W2 + b2         (256 -> 128, fp32 residual add)
  gemm_bf16_kernel<<<dim3(MZ / 16, 1), 256, 0, stream>>>(
      h1, w2T, b2, 256, 128, MZ, /*mode=*/3, /*gelu=*/0, nullptr, z2, z1f);

  // 9) post-LN + channels-first transpose into d_out
  ln_rows_kernel<<<MZ / 8, 256, 0, stream>>>(z2, post_g, post_b, nullptr, out, MZ, 1);
}
